// ResNetBasicblock_22668837388977
// MI455X (gfx1250) — matrix operand order repaired and checked by reading, not compiled, not run on hardware
//
#include <hip/hip_runtime.h>

// ---------------------------------------------------------------------------
// ResNet basic block w/ CGNet gated 3x3 convs on MI455X (gfx1250).
// Implicit GEMM on V_WMMA_F32_16X16X32_BF16, NHWC bf16 activations,
// [tap][O][I] bf16 weights. Weights staged block-cooperatively into LDS
// (double-buffered, async global->LDS) and read back as B-fragments via
// ds_load_b128. Inner 8-step K-block loop fully unrolled so buffer parity,
// yp/yr accumulator routing and prefetch indices are compile-time.
// Border zero-padding via address select against a zeroed scratch page.
// ---------------------------------------------------------------------------

typedef __attribute__((ext_vector_type(16))) __bf16        v16bf;
typedef __attribute__((ext_vector_type(8)))  float          v8f;
typedef __attribute__((ext_vector_type(8)))  unsigned short ushort8;
typedef __attribute__((ext_vector_type(4)))  int            v4i;

typedef __attribute__((address_space(1))) v4i v4i_g;
typedef __attribute__((address_space(3))) v4i v4i_l;

#define BATCH 16
#define CCH   256
#define HH    56
#define WW    56
#define SPAT  (HH * WW)                 // 3136
#define MTOT  (BATCH * SPAT)            // 50176
#define MTILES (MTOT / 16)              // 3136
#define MBLOCKS (MTILES / 8)            // 392
#define NBLOCKS (CCH / 64)              // 4
#define LROW    40                      // padded LDS row stride (ushorts)

#if __has_builtin(__builtin_amdgcn_global_load_async_to_lds_b128) && \
    __has_builtin(__builtin_amdgcn_s_wait_asynccnt)
#define USE_ASYNC 1
#else
#define USE_ASYNC 0
#endif

struct U8x2 { ushort8 lo, hi; };

static __device__ __forceinline__ v16bf frag_of(ushort8 lo, ushort8 hi) {
  U8x2 t{lo, hi};
  return __builtin_bit_cast(v16bf, t);
}

static __device__ __forceinline__ unsigned short f2bf(float f) {
  unsigned int u = __float_as_uint(f);
  u += 0x7FFFu + ((u >> 16) & 1u);      // round-to-nearest-even
  return (unsigned short)(u >> 16);
}

static __device__ __forceinline__ void stage16B(const unsigned short* g,
                                                unsigned short* l) {
#if USE_ASYNC
  __builtin_amdgcn_global_load_async_to_lds_b128((v4i_g*)g, (v4i_l*)l, 0, 0);
#else
  *(ushort8*)l = *(const ushort8*)g;
#endif
}

static __device__ __forceinline__ void wait_stage() {
#if USE_ASYNC
  __builtin_amdgcn_s_wait_asynccnt(0);
#endif
}

// --- zero page for border padding ------------------------------------------
__global__ void zero_pad(unsigned short* z) { z[threadIdx.x] = 0; }

// --- pack x: NCHW fp32 -> NHWC bf16 ----------------------------------------
__global__ void pack_x_bf16(const float* __restrict__ x,
                            unsigned short* __restrict__ xb) {
  size_t i = (size_t)blockIdx.x * 256 + threadIdx.x;
  int    c  = (int)(i & (CCH - 1));
  size_t m  = i >> 8;
  int    ni = (int)(m / SPAT);
  int    sp = (int)(m - (size_t)ni * SPAT);
  xb[i] = f2bf(x[((size_t)(ni * CCH + c)) * SPAT + sp]);
}

// --- pack w: OIHW fp32 -> [tap][O][I] bf16 ---------------------------------
__global__ void pack_w_bf16(const float* __restrict__ w,
                            unsigned short* __restrict__ wb) {
  size_t i = (size_t)blockIdx.x * 256 + threadIdx.x;
  int c   = (int)(i & (CCH - 1));
  int o   = (int)((i >> 8) & (CCH - 1));
  int tap = (int)(i >> 16);
  wb[i] = f2bf(w[((size_t)o * CCH + c) * 9 + tap]);
}

// --- gated conv + BN (+ReLU / +residual), WMMA + LDS-staged weights --------
template <bool FINAL>
__global__ void __launch_bounds__(256)
conv_cg_wmma(const unsigned short* __restrict__ in,    // NHWC bf16 [M, 256]
             const unsigned short* __restrict__ wt,    // [9][256][256] bf16
             const unsigned short* __restrict__ zpad,  // >=48B of zeros
             const float* __restrict__ theta,
             const float* __restrict__ gamma,
             const float* __restrict__ beta,
             const float* __restrict__ mean,
             const float* __restrict__ var,
             const float* __restrict__ xres,           // NCHW fp32 (FINAL)
             unsigned short* __restrict__ out_bf,      // layer-1 output
             float* __restrict__ out_f) {              // layer-2 output
  __shared__ unsigned short sB[2][64 * LROW];          // 2 x 5 KB B tiles

  const int tid  = threadIdx.x;
  const int lane = tid & 31;
  const int wave = tid >> 5;
  const int mb   = blockIdx.x % MBLOCKS;
  const int nb64 = (blockIdx.x / MBLOCKS) * 64;

  const int m_tile = mb * 8 + wave;
  const int half   = lane >> 4;                 // K-half per ISA lane split
  const int l15    = lane & 15;

  // pixel coords for this lane's A-row (M = m_tile*16 + l15)
  const int m_row = m_tile * 16 + l15;
  const int n_img = m_row / SPAT;
  const int sp    = m_row - n_img * SPAT;
  const int oh    = sp / WW;
  const int ow    = sp - oh * WW;

  // staging coords: each thread owns one 16B chunk of the 32x64 B tile
  const int srow = tid >> 2;                    // 0..63 output channel row
  const int scol = (tid & 3) * 8;               // 0,8,16,24 within K-block

  v8f accp[4] = {};
  v8f accr[4] = {};

  auto stage_step = [&](int tap, int cb, int bi) {
    const unsigned short* g =
        wt + ((size_t)(tap * CCH + nb64 + srow)) * CCH + cb * 32 + scol;
    stage16B(g, &sB[bi][srow * LROW + scol]);
  };

  // border handling: select address between pixel and zero page, no branch
  auto load_a = [&](int tap, int cb) -> v16bf {
    const int ky = (tap * 11) >> 5;             // tap/3 for tap in [0,8]
    const int kx = tap - ky * 3;
    const int ih = oh + ky - 1, iw = ow + kx - 1;
    const bool valid = (ih >= 0) && (ih < HH) && (iw >= 0) && (iw < WW);
    const long pix = ((long)n_img * SPAT + ih * WW + iw) * CCH;
    const unsigned short* base = in + pix + cb * 32 + half * 8;
    const ushort8* p = (const ushort8*)(valid ? base : zpad);
    return frag_of(p[0], p[2]);                 // K = h*8.., 16+h*8..
  };

  // ---- software-pipelined K loop: 9 taps x 8 unrolled K-blocks ----
  stage_step(0, 0, 0);
  wait_stage();
  __syncthreads();

  v16bf a_cur = load_a(0, 0);

  for (int tap = 0; tap < 9; ++tap) {
#pragma unroll
    for (int cb = 0; cb < 8; ++cb) {            // compile-time cb
      const int  ntap = (cb == 7) ? tap + 1 : tap;
      const int  ncb  = (cb + 1) & 7;
      const bool more = (cb < 7) || (tap < 8);  // compile-time except cb==7

      if (more) stage_step(ntap, ncb, (cb + 1) & 1);   // async next B tile
      v16bf a_nxt = more ? load_a(ntap, ncb) : a_cur;  // A register prefetch

      const unsigned short* lb = sB[cb & 1];
      v16bf bf4[4];
#pragma unroll
      for (int nt = 0; nt < 4; ++nt) {
        const ushort8* q =
            (const ushort8*)(lb + (nt * 16 + l15) * LROW + half * 8);
        bf4[nt] = frag_of(q[0], q[2]);
      }

      if (cb < 2) {                             // base 64 in-channels -> yp
#pragma unroll
        for (int nt = 0; nt < 4; ++nt)
          accp[nt] = __builtin_amdgcn_wmma_f32_16x16x32_bf16(
              false, a_cur, false, bf4[nt], (short)0, accp[nt], false, false);
      } else {                                  // remaining 192 -> yr
#pragma unroll
        for (int nt = 0; nt < 4; ++nt)
          accr[nt] = __builtin_amdgcn_wmma_f32_16x16x32_bf16(
              false, a_cur, false, bf4[nt], (short)0, accr[nt], false, false);
      }
      a_cur = a_nxt;

      wait_stage();                             // next B tile resident
      __syncthreads();                          // everyone done with current
    }
  }

  // ---- epilogue: gate, BN, (relu | residual+relu) -------------------------
#pragma unroll
  for (int nt = 0; nt < 4; ++nt) {
    const int o  = nb64 + nt * 16 + l15;
    const float th = theta[o];
    const float iv = gamma[o] * rsqrtf(var[o] + 1e-5f);
    const float ad = beta[o] - mean[o] * iv;

#pragma unroll
    for (int i = 0; i < 8; ++i) {
      const int m = m_tile * 16 + i + half * 8; // C/D row per ISA layout
      const float yp = accp[nt][i], yr = accr[nt][i];
      const float d  = 1.f / (1.f + __expf(-2.f * (yp - th)));
      const float y  = (yp + d * yr) * iv + ad;

      if (!FINAL) {
        out_bf[(size_t)m * CCH + o] = f2bf(fmaxf(y, 0.f));
      } else {
        const int ni = m / SPAT;
        const int so = m - ni * SPAT;
        const size_t idx = ((size_t)(ni * CCH + o)) * SPAT + so;
        out_f[idx] = fmaxf(xres[idx] + y, 0.f);
      }
    }
  }
}

extern "C" void kernel_launch(void* const* d_in, const int* in_sizes, int n_in,
                              void* d_out, int out_size, void* d_ws, size_t ws_size,
                              hipStream_t stream) {
  const float* x       = (const float*)d_in[0];
  const float* w_a     = (const float*)d_in[1];
  const float* theta_a = (const float*)d_in[2];
  const float* gamma_a = (const float*)d_in[3];
  const float* beta_a  = (const float*)d_in[4];
  const float* mean_a  = (const float*)d_in[5];
  const float* var_a   = (const float*)d_in[6];
  const float* w_b     = (const float*)d_in[7];
  const float* theta_b = (const float*)d_in[8];
  const float* gamma_b = (const float*)d_in[9];
  const float* beta_b  = (const float*)d_in[10];
  const float* mean_b  = (const float*)d_in[11];
  const float* var_b   = (const float*)d_in[12];

  unsigned short* xb  = (unsigned short*)d_ws;
  unsigned short* hb  = xb + (size_t)MTOT * CCH;
  unsigned short* wab = hb + (size_t)MTOT * CCH;
  unsigned short* wbb = wab + (size_t)9 * CCH * CCH;
  unsigned short* zp  = wbb + (size_t)9 * CCH * CCH;   // 256B zero page

  zero_pad<<<1, 128, 0, stream>>>(zp);
  pack_x_bf16<<<(MTOT * CCH) / 256, 256, 0, stream>>>(x, xb);
  pack_w_bf16<<<(9 * CCH * CCH) / 256, 256, 0, stream>>>(w_a, wab);
  pack_w_bf16<<<(9 * CCH * CCH) / 256, 256, 0, stream>>>(w_b, wbb);

  conv_cg_wmma<false><<<MBLOCKS * NBLOCKS, 256, 0, stream>>>(
      xb, wab, zp, theta_a, gamma_a, beta_a, mean_a, var_a,
      nullptr, hb, nullptr);
  conv_cg_wmma<true><<<MBLOCKS * NBLOCKS, 256, 0, stream>>>(
      hb, wbb, zp, theta_b, gamma_b, beta_b, mean_b, var_b,
      x, nullptr, (float*)d_out);
}
